// RGCN_11424613007387
// MI455X (gfx1250) — matrix operand order repaired and checked by reading, not compiled, not run on hardware
//
#include <hip/hip_runtime.h>
#include <hip/hip_bf16.h>

// ---------------------------------------------------------------------------
// RGCN (2-layer, mean aggregation) for MI455X / gfx1250.
//
// Aggregate-then-transform: messages are linear in x[src] and the mean norm
// is constant per (dst, rel), so
//   A[n, r, :] = sum_{e: dst=n, rel=r} x[src_e, :] / cnt[n, r]  (f32 atomics, L2-resident)
//   out[n]     = A[n] (as [N,512]) @ W[512,Fout] + x[n] @ root + bias
// Dense part runs on v_wmma_f32_16x16x32_bf16 with weights prepacked into
// per-lane B-fragment layout (bf16, one 32B contiguous load per fragment).
// ---------------------------------------------------------------------------

#define NUM_NODES 50000
#define NUM_EDGES 800000
#define NUM_RELS  8
#define FEAT      64                 // IN_F == HID_F == 64
#define KA        (NUM_RELS * FEAT)  // 512 = flattened (rel, feat) K dimension
#define MTILES    (NUM_NODES / 16)   // 3125, exact

typedef __attribute__((ext_vector_type(16))) __bf16 v16bf;
typedef __attribute__((ext_vector_type(8)))  float  v8f;

// ---------------------------------------------------------------------------
// Utility: zero fill
// ---------------------------------------------------------------------------
__global__ void zero_f32(float* __restrict__ p, size_t n) {
  size_t i = (size_t)blockIdx.x * blockDim.x + threadIdx.x;
  size_t stride = (size_t)gridDim.x * blockDim.x;
  for (; i < n; i += stride) p[i] = 0.0f;
}

__global__ void zero_i32(int* __restrict__ p, int n) {
  int i = blockIdx.x * blockDim.x + threadIdx.x;
  if (i < n) p[i] = 0;
}

// ---------------------------------------------------------------------------
// Phase 0: prepack f32 weights [K, FOUT] into bf16 WMMA B-fragments:
//   P[kb][nt][lane][i] = (bf16) W[(kb*32 + (lane>>4)*8 + (i<8 ? i : i+8)) * FOUT + nt*16 + (lane&15)]
// One thread per (kb, nt, lane) writes one 16-element (32 B) fragment.
// ---------------------------------------------------------------------------
__global__ void pack_weights(const float* __restrict__ W,  // [K, FOUT]
                             __bf16* __restrict__ P,       // [K/32][FOUT/16][32][16]
                             int K, int FOUT) {
  int idx = blockIdx.x * blockDim.x + threadIdx.x;
  int NT = FOUT >> 4;
  int total = (K >> 5) * NT * 32;
  if (idx >= total) return;
  int lane = idx & 31;
  int nt   = (idx >> 5) % NT;
  int kb   = idx / (32 * NT);
  int col   = nt * 16 + (lane & 15);
  int kbase = kb * 32 + (lane >> 4) * 8;
  __bf16* dst = P + (size_t)idx * 16;
#pragma unroll
  for (int i = 0; i < 16; ++i)
    dst[i] = (__bf16)W[(size_t)(kbase + (i < 8 ? i : i + 8)) * FOUT + col];
}

// ---------------------------------------------------------------------------
// Phase 1: per-(dst, rel) in-degree counts (shared by both layers)
// ---------------------------------------------------------------------------
__global__ void count_kernel(const long long* __restrict__ edge_index,
                             const long long* __restrict__ edge_type,
                             int* __restrict__ cnt) {
  int e = blockIdx.x * blockDim.x + threadIdx.x;
  if (e >= NUM_EDGES) return;
  int dst = (int)edge_index[NUM_EDGES + e];
  int rel = (int)edge_type[e];
  atomicAdd(&cnt[dst * NUM_RELS + rel], 1);
}

// ---------------------------------------------------------------------------
// Phase 2: scatter normalized raw features into A[n, rel, f]  (f32 atomics)
// One thread per (edge, feature). A fits in L2 (102.4 MB < 192 MB).
// ---------------------------------------------------------------------------
__global__ void scatter_kernel(const long long* __restrict__ edge_index,
                               const long long* __restrict__ edge_type,
                               const int* __restrict__ cnt,
                               const float* __restrict__ feat,   // [N, 64]
                               float* __restrict__ A) {          // [N, 512]
  size_t g = (size_t)blockIdx.x * blockDim.x + threadIdx.x;
  size_t e = g >> 6;           // edge id
  int    f = (int)(g & 63);    // feature id
  if (e >= NUM_EDGES) return;
  int src = (int)edge_index[e];
  int dst = (int)edge_index[NUM_EDGES + e];
  int rel = (int)edge_type[e];
  int c = cnt[dst * NUM_RELS + rel];
  float norm = 1.0f / (float)(c < 1 ? 1 : c);
  float v = feat[(size_t)src * FEAT + f] * norm;
  atomicAdd(&A[((size_t)dst * NUM_RELS + rel) * FEAT + f], v);
}

// ---------------------------------------------------------------------------
// Phase 3: dense transform via WMMA bf16 (f32 accumulate).
//   out[N, FOUT] = A[N, 512] @ W + X[N, 64] @ RW + bias
// One wave32 computes one 16-row M-tile x full FOUT.
//   A-fragments: 4x b128 f32 loads + packed cvt to bf16 in registers.
//   B-fragments: one contiguous 32B bf16 load from the prepacked buffer.
//
// Fragment layouts (CDNA5 ISA §7.12.2, wave32):
//   A 16x32 bf16 : lane l -> row (l&15); elems 0..7  = K[(l>>4)*8 .. +7],
//                                         elems 8..15 = K[16+(l>>4)*8 .. +7]
//   B 32x16 bf16 : lane l -> col (l&15); elem i = K[(l>>4)*8 + i + (i>=8 ? 8 : 0)]
//   C/D 16x16 f32: lane l -> col (l&15); elem i = row (i + 8*(l>>4))
// ---------------------------------------------------------------------------
__device__ __forceinline__ void cvt4(const float4 u, v16bf& a, int o) {
  a[o + 0] = (__bf16)u.x; a[o + 1] = (__bf16)u.y;
  a[o + 2] = (__bf16)u.z; a[o + 3] = (__bf16)u.w;
}

__device__ __forceinline__ v16bf load_a_frag(const float* __restrict__ base) {
  // base already points at row start + kblock + (lane>>4)*8
  float4 u0 = *reinterpret_cast<const float4*>(base);
  float4 u1 = *reinterpret_cast<const float4*>(base + 4);
  float4 u2 = *reinterpret_cast<const float4*>(base + 16);
  float4 u3 = *reinterpret_cast<const float4*>(base + 20);
  v16bf a;
  cvt4(u0, a, 0); cvt4(u1, a, 4); cvt4(u2, a, 8); cvt4(u3, a, 12);
  return a;
}

template <int FOUT>
__global__ void __launch_bounds__(256)
rgcn_gemm(const float*  __restrict__ A,     // [N, 512]
          const float*  __restrict__ X,     // [N, 64]  (root-path input)
          const __bf16* __restrict__ Pw,    // packed [512/32][NT][32][16]
          const __bf16* __restrict__ Pr,    // packed [64/32][NT][32][16]
          const float*  __restrict__ bias,  // [FOUT]
          float*        __restrict__ out,   // [N, FOUT]
          int relu) {
  constexpr int NT = FOUT / 16;
  const int lane  = threadIdx.x & 31;
  const int wave  = threadIdx.x >> 5;
  const int mtile = blockIdx.x * 8 + wave;
  if (mtile >= MTILES) return;   // wave-uniform: EXEC stays all-ones for WMMA

  const int mbase = mtile * 16;
  const int arow  = mbase + (lane & 15);
  const int kh8   = (lane >> 4) * 8;
  const int col0  = lane & 15;

  v8f acc[NT];
#pragma unroll
  for (int t = 0; t < NT; ++t) acc[t] = (v8f){0.f,0.f,0.f,0.f,0.f,0.f,0.f,0.f};

  // ---- relation part: K = 512 over A / Pw ----
  const float* arowp = A + (size_t)arow * KA + kh8;
#pragma unroll 4
  for (int kbi = 0; kbi < KA / 32; ++kbi) {
    v16bf afrag = load_a_frag(arowp + kbi * 32);
#pragma unroll
    for (int t = 0; t < NT; ++t) {
      v16bf bfrag = *reinterpret_cast<const v16bf*>(
          Pw + (((size_t)kbi * NT + t) * 32 + lane) * 16);
      acc[t] = __builtin_amdgcn_wmma_f32_16x16x32_bf16(
          false, afrag, false, bfrag, (short)0, acc[t], false, false);
    }
  }

  // ---- root part: K = 64 over X / Pr ----
  const float* xrowp = X + (size_t)arow * FEAT + kh8;
#pragma unroll
  for (int kbi = 0; kbi < FEAT / 32; ++kbi) {
    v16bf afrag = load_a_frag(xrowp + kbi * 32);
#pragma unroll
    for (int t = 0; t < NT; ++t) {
      v16bf bfrag = *reinterpret_cast<const v16bf*>(
          Pr + (((size_t)kbi * NT + t) * 32 + lane) * 16);
      acc[t] = __builtin_amdgcn_wmma_f32_16x16x32_bf16(
          false, afrag, false, bfrag, (short)0, acc[t], false, false);
    }
  }

  // ---- epilogue: + bias, optional ReLU, store ----
  const int rh = (lane >> 4) * 8;
#pragma unroll
  for (int t = 0; t < NT; ++t) {
    const int col = t * 16 + col0;
    const float bv = bias[col];
#pragma unroll
    for (int i = 0; i < 8; ++i) {
      float v = acc[t][i] + bv;
      if (relu) v = fmaxf(v, 0.0f);
      out[(size_t)(mbase + rh + i) * FOUT + col] = v;
    }
  }
}

// ---------------------------------------------------------------------------
// Host-side launch. Workspace layout (offsets in bytes):
//   [0)        cnt : int32 [N*8]                      1.6 MB
//   [2 MB)     A   : f32  [N*512]                   102.4 MB
//   [+A]       h1  : f32  [N*64]                     12.8 MB
//   [+h1]      Pw1 : bf16 [16][4][32][16]            64 KB
//   [+Pw1]     Pr1 : bf16 [ 2][4][32][16]             8 KB
//   [+Pr1]     Pw2 : bf16 [16][2][32][16]            32 KB
//   [+Pw2]     Pr2 : bf16 [ 2][2][32][16]             4 KB
// ---------------------------------------------------------------------------
extern "C" void kernel_launch(void* const* d_in, const int* in_sizes, int n_in,
                              void* d_out, int out_size, void* d_ws, size_t ws_size,
                              hipStream_t stream) {
  const float*     x     = (const float*)d_in[0];
  const long long* ei    = (const long long*)d_in[1];   // int64 [2, E]
  const long long* et    = (const long long*)d_in[2];   // int64 [E]
  const float*     W1    = (const float*)d_in[3];       // [8,64,64] == [512,64]
  const float*     root1 = (const float*)d_in[4];       // [64,64]
  const float*     b1    = (const float*)d_in[5];       // [64]
  const float*     W2    = (const float*)d_in[6];       // [8,64,32] == [512,32]
  const float*     root2 = (const float*)d_in[7];       // [64,32]
  const float*     b2    = (const float*)d_in[8];       // [32]
  float*           out   = (float*)d_out;               // [N, 32]

  char* ws = (char*)d_ws;
  int*    cnt = (int*)ws;
  float*  A   = (float*)(ws + (size_t)2 * 1024 * 1024);
  float*  h1  = (float*)((char*)A + (size_t)NUM_NODES * KA * sizeof(float));
  __bf16* Pw1 = (__bf16*)((char*)h1 + (size_t)NUM_NODES * FEAT * sizeof(float));
  __bf16* Pr1 = Pw1 + (size_t)(KA / 32) * 4 * 32 * 16;
  __bf16* Pw2 = Pr1 + (size_t)(FEAT / 32) * 4 * 32 * 16;
  __bf16* Pr2 = Pw2 + (size_t)(KA / 32) * 2 * 32 * 16;

  const size_t nA = (size_t)NUM_NODES * KA;   // 25.6M floats

  // weight prepack (independent of graph data)
  pack_weights<<<(16 * 4 * 32 + 255) / 256, 256, 0, stream>>>(W1,    Pw1, KA,   64);
  pack_weights<<<( 2 * 4 * 32 + 255) / 256, 256, 0, stream>>>(root1, Pr1, FEAT, 64);
  pack_weights<<<(16 * 2 * 32 + 255) / 256, 256, 0, stream>>>(W2,    Pw2, KA,   32);
  pack_weights<<<( 2 * 2 * 32 + 255) / 256, 256, 0, stream>>>(root2, Pr2, FEAT, 32);

  // counts (shared by both layers)
  zero_i32<<<(NUM_NODES * NUM_RELS + 255) / 256, 256, 0, stream>>>(cnt, NUM_NODES * NUM_RELS);
  count_kernel<<<(NUM_EDGES + 255) / 256, 256, 0, stream>>>(ei, et, cnt);

  // ---- layer 1 ----
  zero_f32<<<2048, 256, 0, stream>>>(A, nA);
  scatter_kernel<<<(unsigned)(((size_t)NUM_EDGES * FEAT) / 256), 256, 0, stream>>>(ei, et, cnt, x, A);
  rgcn_gemm<64><<<(MTILES + 7) / 8, 256, 0, stream>>>(A, x, Pw1, Pr1, b1, h1, /*relu=*/1);

  // ---- layer 2 ----
  zero_f32<<<2048, 256, 0, stream>>>(A, nA);
  scatter_kernel<<<(unsigned)(((size_t)NUM_EDGES * FEAT) / 256), 256, 0, stream>>>(ei, et, cnt, h1, A);
  rgcn_gemm<32><<<(MTILES + 7) / 8, 256, 0, stream>>>(A, h1, Pw2, Pr2, b2, out, /*relu=*/0);
}
